// AttentionBlock_16509854286658
// MI455X (gfx1250) — matrix operand order repaired and checked by reading, not compiled, not run on hardware
//
#include <hip/hip_runtime.h>
#include <hip/hip_bf16.h>
#include <math.h>

typedef __bf16 bf16_t;
typedef __attribute__((ext_vector_type(16))) __bf16 v16bf;
typedef __attribute__((ext_vector_type(8)))  __bf16 v8bf;
typedef __attribute__((ext_vector_type(8)))  float  v8f;
typedef __attribute__((ext_vector_type(4)))  int    v4i_t;

#define B_    8
#define C_    256
#define HW_   4096        // N tokens per batch (64*64)
#define G_    8
#define CG_   32          // channels per group
#define EPS_  1e-5f
#define SCALE_ 0.0625f    // 256^-0.5

// ---- gfx1250 async global->LDS path (ASYNCcnt), guarded so we always compile
#if defined(__has_builtin)
#if __has_builtin(__builtin_amdgcn_global_load_async_to_lds_b128) && \
    __has_builtin(__builtin_amdgcn_s_wait_asynccnt)
#define ASYNC_LDS 1
#endif
#endif
#ifndef ASYNC_LDS
#define ASYNC_LDS 0
#endif

static __device__ inline v16bf cat8(v8bf lo, v8bf hi) {
  v16bf r;
#pragma unroll
  for (int i = 0; i < 8; ++i) { r[i] = lo[i]; r[8 + i] = hi[i]; }
  return r;
}

// ---------------------------------------------------------------------------
// 1) GroupNorm: x[B,C,H,W] -> t (channel-last tokens [B,N,C]) fp32 + bf16
// ---------------------------------------------------------------------------
__global__ void k_groupnorm(const float* __restrict__ x,
                            const float* __restrict__ gamma,
                            const float* __restrict__ beta,
                            float* __restrict__ tf32,
                            bf16_t* __restrict__ tbf) {
  __shared__ float s_sum[256], s_sq[256];
  __shared__ float s_mean, s_rstd;
  int bg = blockIdx.x;
  int b = bg / G_, g = bg % G_;
  int tid = threadIdx.x;
  const float* xb = x + ((size_t)b * C_ + g * CG_) * HW_;

  float sum = 0.f, sq = 0.f;
  for (int idx = tid; idx < CG_ * HW_; idx += 256) {
    int c = idx & (CG_ - 1);
    int s = idx >> 5;
    float v = xb[(size_t)c * HW_ + s];
    sum += v; sq += v * v;
  }
  s_sum[tid] = sum; s_sq[tid] = sq;
  __syncthreads();
  for (int off = 128; off > 0; off >>= 1) {
    if (tid < off) { s_sum[tid] += s_sum[tid + off]; s_sq[tid] += s_sq[tid + off]; }
    __syncthreads();
  }
  if (tid == 0) {
    float mean = s_sum[0] / (float)(CG_ * HW_);
    float var  = s_sq[0] / (float)(CG_ * HW_) - mean * mean;
    s_mean = mean;
    s_rstd = rsqrtf(var + EPS_);
  }
  __syncthreads();
  float mean = s_mean, rstd = s_rstd;

  for (int idx = tid; idx < CG_ * HW_; idx += 256) {
    int c = idx & (CG_ - 1);
    int s = idx >> 5;
    int ch = g * CG_ + c;
    float v = xb[(size_t)c * HW_ + s];
    float y = (v - mean) * rstd * gamma[ch] + beta[ch];
    size_t o = ((size_t)b * HW_ + s) * C_ + ch;
    tf32[o] = y;
    tbf[o]  = (bf16_t)y;
  }
}

// ---------------------------------------------------------------------------
// 2) Convert the four 256x256 weight matrices to bf16
// ---------------------------------------------------------------------------
__global__ void k_cvt_w(const float* __restrict__ wq, const float* __restrict__ wk,
                        const float* __restrict__ wv, const float* __restrict__ wp,
                        bf16_t* __restrict__ wbf) {
  int i = blockIdx.x * blockDim.x + threadIdx.x;
  const int n = C_ * C_;
  if (i < n) {
    wbf[0 * n + i] = (bf16_t)wq[i];
    wbf[1 * n + i] = (bf16_t)wk[i];
    wbf[2 * n + i] = (bf16_t)wv[i];
    wbf[3 * n + i] = (bf16_t)wp[i];
  }
}

// ---------------------------------------------------------------------------
// 3) QKV GEMM: Y = t @ W^T + bias, bf16 in / bf16 out, f32 accumulate.
// One wave per 16x16 tile; K=256 -> 8x wmma_f32_16x16x32_bf16.
// ---------------------------------------------------------------------------
__global__ void k_gemm_qkv(const bf16_t* __restrict__ tbf,
                           const bf16_t* __restrict__ wbf,
                           const float* __restrict__ bq,
                           const float* __restrict__ bk,
                           const float* __restrict__ bv,
                           bf16_t* __restrict__ qkv) {
  int sel = blockIdx.z;
  const bf16_t* W  = wbf + (size_t)sel * C_ * C_;
  const float* bias = (sel == 0) ? bq : (sel == 1) ? bk : bv;
  bf16_t* out = qkv + (size_t)sel * B_ * HW_ * C_;

  int wave  = (blockIdx.x * blockDim.x + threadIdx.x) >> 5;
  int lane  = threadIdx.x & 31;
  int lmod  = lane & 15, lhalf = lane >> 4;
  int rowTile = wave >> 4;      // [0, 2048)
  int colTile = wave & 15;      // [0, 16)
  int row0 = rowTile * 16, col0 = colTile * 16;

  const bf16_t* arow = tbf + (size_t)(row0 + lmod) * C_;
  const bf16_t* brow = W   + (size_t)(col0 + lmod) * C_;
  v8f acc = {};
#pragma unroll
  for (int k0 = 0; k0 < C_; k0 += 32) {
    v16bf a  = cat8(*(const v8bf*)(arow + k0 + lhalf * 8),
                    *(const v8bf*)(arow + k0 + 16 + lhalf * 8));
    v16bf bm = cat8(*(const v8bf*)(brow + k0 + lhalf * 8),
                    *(const v8bf*)(brow + k0 + 16 + lhalf * 8));
    acc = __builtin_amdgcn_wmma_f32_16x16x32_bf16(false, a, false, bm,
                                                  (short)0, acc, false, false);
  }
  float bb = bias[col0 + lmod];
#pragma unroll
  for (int r = 0; r < 8; ++r) {
    int row = row0 + r + lhalf * 8;
    out[(size_t)row * C_ + col0 + lmod] = (bf16_t)(acc[r] + bb);
  }
}

// ---------------------------------------------------------------------------
// 4) Flash attention, cooperative within a block:
//    - 4 waves = 4 consecutive 16-query tiles of the SAME batch
//    - each 32-key chunk of K and V (16KB each) is staged to LDS once per
//      block via GLOBAL_LOAD_ASYNC_TO_LDS_B128 (ASYNCcnt) when available,
//      then all waves read fragments from LDS (contiguous ds_load for K,
//      strided transpose-gather for V resolved in LDS instead of HBM)
//    - online softmax via wave shuffles; P transposed D->A layout through a
//      per-wave LDS scratch (same-wave DS ordering, no barrier needed)
// ---------------------------------------------------------------------------
__global__ void k_attn(const bf16_t* __restrict__ qbf,
                       const bf16_t* __restrict__ kbf,
                       const bf16_t* __restrict__ vbf,
                       bf16_t* __restrict__ obf) {
  __shared__ alignas(16) bf16_t klds[32 * C_];     // [key][c], 16KB
  __shared__ alignas(16) bf16_t vlds[32 * C_];     // [key][c], 16KB
  __shared__ alignas(16) bf16_t pbuf[4][16 * 32];  // per-wave P transpose
  int wib  = threadIdx.x >> 5;
  int lane = threadIdx.x & 31;
  int lmod = lane & 15, lhalf = lane >> 4;
  int tile = blockIdx.x * 4 + wib;   // 4 query tiles per block, same batch
  int b  = tile >> 8;                // 256 query tiles per batch
  int qt = tile & 255;
  size_t base = (size_t)b * HW_ * C_;
  int row0 = qt * 16;

  // Q fragments (A-layout), 8 chunks of K=32, held in registers
  v16bf qf[8];
  const bf16_t* qrow = qbf + base + (size_t)(row0 + lmod) * C_;
#pragma unroll
  for (int kc = 0; kc < 8; ++kc) {
    int k0 = kc * 32;
    qf[kc] = cat8(*(const v8bf*)(qrow + k0 + lhalf * 16),
                  *(const v8bf*)(qrow + k0 + lhalf * 16 + 8));
  }

  v8f o[16] = {};
  float m[8], l[8];
#pragma unroll
  for (int r = 0; r < 8; ++r) { m[r] = -3.0e38f; l[r] = 0.f; }
  bf16_t* pb = &pbuf[wib][0];

  for (int j0 = 0; j0 < HW_; j0 += 32) {
    // ---- cooperative stage of K,V 32-key tiles into LDS ----
    const bf16_t* kg = kbf + base + (size_t)j0 * C_;
    const bf16_t* vg = vbf + base + (size_t)j0 * C_;
    __syncthreads();   // previous chunk's LDS reads complete
    if (j0 + 32 < HW_) {
      __builtin_prefetch(kg + 32 * C_, 0, 0);   // global_prefetch_b8
      __builtin_prefetch(vg + 32 * C_, 0, 0);
    }
#if ASYNC_LDS
    for (int i = threadIdx.x; i < (32 * C_) / 8; i += 128) {
      __builtin_amdgcn_global_load_async_to_lds_b128(
          (v4i_t*)(kg + (size_t)i * 8), (v4i_t*)(klds + i * 8), 0, 0);
      __builtin_amdgcn_global_load_async_to_lds_b128(
          (v4i_t*)(vg + (size_t)i * 8), (v4i_t*)(vlds + i * 8), 0, 0);
    }
    __builtin_amdgcn_s_wait_asynccnt(0);
#else
    for (int i = threadIdx.x; i < (32 * C_) / 8; i += 128) {
      ((v8bf*)klds)[i] = ((const v8bf*)kg)[i];
      ((v8bf*)vlds)[i] = ((const v8bf*)vg)[i];
    }
#endif
    __syncthreads();

    // ---- S = Q K^T for 32 keys (two 16-key fragments), K from LDS ----
    v8f s0 = {}, s1 = {};
    const bf16_t* krow0 = klds + (size_t)lmod * C_ + lhalf * 16;
    const bf16_t* krow1 = krow0 + 16 * C_;
#pragma unroll
    for (int kc = 0; kc < 8; ++kc) {
      int c0 = kc * 32;
      v16bf b0 = cat8(*(const v8bf*)(krow0 + c0), *(const v8bf*)(krow0 + c0 + 8));
      v16bf b1 = cat8(*(const v8bf*)(krow1 + c0), *(const v8bf*)(krow1 + c0 + 8));
      s0 = __builtin_amdgcn_wmma_f32_16x16x32_bf16(false, qf[kc], false, b0,
                                                   (short)0, s0, false, false);
      s1 = __builtin_amdgcn_wmma_f32_16x16x32_bf16(false, qf[kc], false, b1,
                                                   (short)0, s1, false, false);
    }
#pragma unroll
    for (int r = 0; r < 8; ++r) { s0[r] *= SCALE_; s1[r] *= SCALE_; }

    // ---- online softmax (row stats across 16-lane groups) ----
    float alpha[8];
#pragma unroll
    for (int r = 0; r < 8; ++r) {
      float sm = fmaxf(s0[r], s1[r]);
#pragma unroll
      for (int off = 8; off >= 1; off >>= 1)
        sm = fmaxf(sm, __shfl_xor(sm, off, 32));
      float mn = fmaxf(m[r], sm);
      alpha[r] = __expf(m[r] - mn);
      m[r] = mn;
      float p0 = __expf(s0[r] - mn);
      float p1 = __expf(s1[r] - mn);
      s0[r] = p0; s1[r] = p1;
      float rs = p0 + p1;
#pragma unroll
      for (int off = 8; off >= 1; off >>= 1)
        rs += __shfl_xor(rs, off, 32);
      l[r] = l[r] * alpha[r] + rs;
    }
#pragma unroll
    for (int cc = 0; cc < 16; ++cc)
#pragma unroll
      for (int r = 0; r < 8; ++r) o[cc][r] *= alpha[r];

    // ---- transpose P: D-layout -> (row,key) LDS -> A-layout (same wave) ----
#pragma unroll
    for (int r = 0; r < 8; ++r) {
      int row = r + lhalf * 8;
      pb[row * 32 + lmod]      = (bf16_t)s0[r];
      pb[row * 32 + 16 + lmod] = (bf16_t)s1[r];
    }
    v16bf pa = cat8(*(const v8bf*)(pb + lmod * 32 + lhalf * 16),
                    *(const v8bf*)(pb + lmod * 32 + lhalf * 16 + 8));

    // ---- O += P @ V  (16 output-channel chunks, K=32 keys), V from LDS ----
    const bf16_t* vrow = vlds + (size_t)(lhalf * 16) * C_ + lmod;
#pragma unroll
    for (int cc = 0; cc < 16; ++cc) {
      v16bf bv_;
#pragma unroll
      for (int e = 0; e < 16; ++e) bv_[e] = vrow[(size_t)e * C_ + cc * 16];
      o[cc] = __builtin_amdgcn_wmma_f32_16x16x32_bf16(false, pa, false, bv_,
                                                      (short)0, o[cc], false, false);
    }
  }

  // ---- epilogue: O /= l, store bf16 ----
#pragma unroll
  for (int r = 0; r < 8; ++r) l[r] = 1.0f / l[r];
  bf16_t* orow = obf + base + (size_t)row0 * C_;
#pragma unroll
  for (int cc = 0; cc < 16; ++cc)
#pragma unroll
    for (int r = 0; r < 8; ++r) {
      int row = r + lhalf * 8;
      orow[(size_t)row * C_ + cc * 16 + lmod] = (bf16_t)(o[cc][r] * l[r]);
    }
}

// ---------------------------------------------------------------------------
// 5) Output projection + residual, scatter back to NCHW fp32
// ---------------------------------------------------------------------------
__global__ void k_proj(const bf16_t* __restrict__ obf,
                       const bf16_t* __restrict__ wpbf,
                       const float* __restrict__ bp,
                       const float* __restrict__ tf32,
                       float* __restrict__ out) {
  int wave = (blockIdx.x * blockDim.x + threadIdx.x) >> 5;
  int lane = threadIdx.x & 31;
  int lmod = lane & 15, lhalf = lane >> 4;
  int rowTile = wave >> 4;
  int colTile = wave & 15;
  int row0 = rowTile * 16, col0 = colTile * 16;

  const bf16_t* arow = obf  + (size_t)(row0 + lmod) * C_;
  const bf16_t* brow = wpbf + (size_t)(col0 + lmod) * C_;
  v8f acc = {};
#pragma unroll
  for (int k0 = 0; k0 < C_; k0 += 32) {
    v16bf a  = cat8(*(const v8bf*)(arow + k0 + lhalf * 8),
                    *(const v8bf*)(arow + k0 + 16 + lhalf * 8));
    v16bf bm = cat8(*(const v8bf*)(brow + k0 + lhalf * 8),
                    *(const v8bf*)(brow + k0 + 16 + lhalf * 8));
    acc = __builtin_amdgcn_wmma_f32_16x16x32_bf16(false, a, false, bm,
                                                  (short)0, acc, false, false);
  }
  int col = col0 + lmod;
  float bias = bp[col];
#pragma unroll
  for (int r = 0; r < 8; ++r) {
    int row = row0 + r + lhalf * 8;       // global token row in [0, B*N)
    int b = row >> 12;                    // / 4096
    int s = row & 4095;
    float val = acc[r] + bias + tf32[(size_t)row * C_ + col];
    out[((size_t)b * C_ + col) * HW_ + s] = val;
  }
}

// ---------------------------------------------------------------------------
extern "C" void kernel_launch(void* const* d_in, const int* in_sizes, int n_in,
                              void* d_out, int out_size, void* d_ws, size_t ws_size,
                              hipStream_t stream) {
  const float* x     = (const float*)d_in[0];
  const float* gamma = (const float*)d_in[1];
  const float* beta  = (const float*)d_in[2];
  const float* Wq = (const float*)d_in[3];
  const float* bq = (const float*)d_in[4];
  const float* Wk = (const float*)d_in[5];
  const float* bk = (const float*)d_in[6];
  const float* Wv = (const float*)d_in[7];
  const float* bv = (const float*)d_in[8];
  const float* Wp = (const float*)d_in[9];
  const float* bp = (const float*)d_in[10];
  float* out = (float*)d_out;

  // workspace layout (bytes)
  char* ws = (char*)d_ws;
  float*  tf32 = (float*) (ws);                 // 33,554,432 B
  bf16_t* tbf  = (bf16_t*)(ws + 33554432);      // 16,777,216 B
  bf16_t* qkv  = (bf16_t*)(ws + 50331648);      // 3 x 16,777,216 B
  bf16_t* obf  = (bf16_t*)(ws + 100663296);     // 16,777,216 B
  bf16_t* wbf  = (bf16_t*)(ws + 117440512);     // 524,288 B

  k_cvt_w    <<<dim3(256),        dim3(256), 0, stream>>>(Wq, Wk, Wv, Wp, wbf);
  k_groupnorm<<<dim3(B_ * G_),    dim3(256), 0, stream>>>(x, gamma, beta, tf32, tbf);
  k_gemm_qkv <<<dim3(8192, 1, 3), dim3(128), 0, stream>>>(tbf, wbf, bq, bk, bv, qkv);
  k_attn     <<<dim3(512),        dim3(128), 0, stream>>>(qkv,
                                                          qkv + (size_t)1 * B_ * HW_ * C_,
                                                          qkv + (size_t)2 * B_ * HW_ * C_,
                                                          obf);
  k_proj     <<<dim3(8192),       dim3(128), 0, stream>>>(obf, wbf + 3 * C_ * C_, bp, tf32, out);
}
